// Net_74431783239994
// MI455X (gfx1250) — matrix operand order repaired and checked by reading, not compiled, not run on hardware
//
#include <hip/hip_runtime.h>
#include <hip/hip_bf16.h>
#include <stdint.h>

typedef __attribute__((ext_vector_type(16))) _Float16     v16h;
typedef __attribute__((ext_vector_type(8)))  _Float16     v8h;
typedef __attribute__((ext_vector_type(8)))  float        v8f;
typedef __attribute__((ext_vector_type(4)))  unsigned int u32x4;
typedef __attribute__((ext_vector_type(8)))  int          i32x8;
typedef __attribute__((ext_vector_type(4)))  int          i32x4;

#define BATCH   65536
#define MBLK    128      // rows per workgroup (8 waves x 16-row M-tiles)
#define ACT_STR 264      // LDS activation stride in halfs (bank-conflict free)
#define WT_STRH 40       // LDS weight-tile stride in halfs (32 + 8 pad via TDM pad fields)

#define USE_TDM 1

#define NLAYERS 20
__device__ __constant__ const int kSizes[21] =
    {784, 256, 256, 128, 64, 32, 16, 8, 4, 2, 2, 2, 4, 8, 16, 32, 64, 128, 256, 256, 256};

struct Params  { const _Float16* W16[20]; const float* Bv[20]; };
struct PrepPar { const float* W[20]; _Float16* D[20]; };

__device__ __forceinline__ uint32_t mixh(uint32_t x) {
    x ^= x >> 16; x *= 0x7feb352du;
    x ^= x >> 15; x *= 0x846ca68bu;
    x ^= x >> 16; return x;
}

// ---------------------------------------------------------------------------
// Kernel P: convert + zero-pad all weights to f16 in WMMA-tile-friendly layout
//           layer L: [Npad x Kpad], Npad = ceil16(NOUT), Kpad = ceil32(NIN)
// Also zeroes the BN-stat accumulators (block 20).
// ---------------------------------------------------------------------------
__global__ void k_prep(PrepPar q, float* __restrict__ gsum, float* __restrict__ gsq) {
    const int L = blockIdx.x;
    if (L == NLAYERS) {                      // stats zero block
        gsum[threadIdx.x] = 0.f;
        gsq[threadIdx.x]  = 0.f;
        return;
    }
    const int nin  = kSizes[L];
    const int nout = kSizes[L + 1];
    const int kp   = ((nin  + 31) >> 5) << 5;
    const int np   = ((nout + 15) >> 4) << 4;
    const float* __restrict__ W = q.W[L];
    _Float16*    __restrict__ D = q.D[L];
    const int total = np * kp;
    for (int idx = threadIdx.x; idx < total; idx += blockDim.x) {
        const int n = idx / kp;
        const int k = idx - n * kp;
        D[idx] = (n < nout && k < nin) ? (_Float16)W[(size_t)n * nin + k] : (_Float16)0.f;
    }
}

// ---------------------------------------------------------------------------
// Kernel 1: h1 = relu(x @ W1^T + b1), f16 out; per-column sum / sumsq for BN.
// One workgroup = 128 rows x 256 cols, K loop over 800 (padded) in 32-chunks.
// f16 weight slabs staged into LDS by the TDM, double-buffered; B fragments
// pipelined 4-deep via a register queue with in-place reload (no rotating
// copies, so the allocator cannot coalesce the queue slots).
// ---------------------------------------------------------------------------
__global__ __launch_bounds__(256) void k_fc1(const float* __restrict__ x,
                                             const _Float16* __restrict__ W1h, // 256 x 800
                                             const float* __restrict__ b1,
                                             _Float16* __restrict__ h1,
                                             float* __restrict__ gsum,
                                             float* __restrict__ gsq) {
    __shared__ _Float16 wt[2][256 * WT_STRH];   // 2 x 20 KB, stride 40 halfs
    __shared__ float csum[256], csq[256];

    const int tid  = threadIdx.x;
    const int lane = tid & 31;
    const int wv   = tid >> 5;
    const int lo   = (lane < 16) ? 1 : 0;
    const int lidx = lane & 15;

    if (tid < 256) { csum[tid] = 0.f; csq[tid] = 0.f; }

    const int    rowBase = blockIdx.x * MBLK + wv * 16;   // this wave's 16 rows
    const float* xr      = x + (size_t)(rowBase + lidx) * 784;

    const v8f zerov = {0.f, 0.f, 0.f, 0.f, 0.f, 0.f, 0.f, 0.f};
    v8f acc[16];
    #pragma unroll
    for (int nt = 0; nt < 16; ++nt) acc[nt] = zerov;

#if USE_TDM && __has_builtin(__builtin_amdgcn_tensor_load_to_lds)
    #define FC1_TDM 1
#else
    #define FC1_TDM 0
#endif

#if FC1_TDM
    // Issue a TDM load of W1h[:, k0..k0+31] (f16) into wt[buf], padded to
    // stride 40 halfs (pad 4 DWORDs every 16 DWORDs).
    auto tdm_issue = [&](int buf, int k0) {
        const uint32_t ldsaddr = (uint32_t)(uintptr_t)(&wt[buf][0]);
        const uint64_t ga      = (uint64_t)(uintptr_t)W1h + (uint64_t)k0 * 2u;
        u32x4 g0;
        g0[0] = 1u;                                   // count=1, user descriptor
        g0[1] = ldsaddr;                              // lds_addr
        g0[2] = (uint32_t)ga;                         // global_addr lo
        g0[3] = ((uint32_t)(ga >> 32) & 0x01FFFFFFu) | (2u << 30); // addr hi | type=2
        i32x8 g1;
        g1[0] = (int)((1u << 16)        // data_size = 2B
                    | (1u << 20)        // pad_enable
                    | (3u << 22)        // pad_interval: every 16 DWORDs (64B)
                    | (3u << 25));      // pad_amount: 4 DWORDs (16B) -> stride 40 halfs
        g1[1] = (int)((800u & 0xFFFFu) << 16);        // tensor_dim0 = 800 (pre-padded)
        g1[2] = (int)(256u << 16);                    // dim0 hi=0 | tensor_dim1 = 256
        g1[3] = (int)(32u << 16);                     // tile_dim0 = 32
        g1[4] = 256;                                  // tile_dim1 = 256
        g1[5] = 800;                                  // tensor_dim0_stride = 800
        g1[6] = 0;
        g1[7] = 0;
        const i32x4 gz4 = {0, 0, 0, 0};
        const i32x8 gz8 = {0, 0, 0, 0, 0, 0, 0, 0};
        __builtin_amdgcn_tensor_load_to_lds(g0, g1, gz4, gz4, gz8, 0);
    };
    if (wv == 0) tdm_issue(0, 0);
#endif

    for (int it = 0; it < 25; ++it) {
        const int k0  = it * 32;
        const int cur = it & 1;

#if FC1_TDM
        if (wv == 0) __builtin_amdgcn_s_wait_tensorcnt(0);
        __syncthreads();                         // wt[cur] ready; prior reads of it done
        if (wv == 0 && it + 1 < 25) tdm_issue(cur ^ 1, k0 + 32);   // overlap next DMA
#else
        __syncthreads();
        {   // manual cooperative stage: thread t copies row n=t, 32 halfs (64B)
            const int n = tid;
            const v8h* srcw = (const v8h*)(W1h + (size_t)n * 800 + k0);
            v8h w0 = srcw[0], w1 = srcw[1], w2 = srcw[2], w3 = srcw[3];
            v8h* dstw = (v8h*)(&wt[cur][n * WT_STRH]);
            dstw[0] = w0; dstw[1] = w1; dstw[2] = w2; dstw[3] = w3;
        }
#endif
        // A fragment (16x32 f16) from x (f32 -> f16 convert; x is f32 in HBM).
        // Lane layout per ISA: lo lanes K in {0-7,16-23}, hi lanes K in {8-15,24-31}.
        const int koff = k0 + (lo ? 0 : 8);
        v16h a;
        if (k0 + 32 <= 784) {
            const float4* p0 = (const float4*)(xr + koff);
            const float4* p1 = (const float4*)(xr + koff + 16);
            const float4 f0 = p0[0], f1 = p0[1], f2 = p1[0], f3 = p1[1];
            a[0]  = (_Float16)f0.x; a[1]  = (_Float16)f0.y; a[2]  = (_Float16)f0.z; a[3]  = (_Float16)f0.w;
            a[4]  = (_Float16)f1.x; a[5]  = (_Float16)f1.y; a[6]  = (_Float16)f1.z; a[7]  = (_Float16)f1.w;
            a[8]  = (_Float16)f2.x; a[9]  = (_Float16)f2.y; a[10] = (_Float16)f2.z; a[11] = (_Float16)f2.w;
            a[12] = (_Float16)f3.x; a[13] = (_Float16)f3.y; a[14] = (_Float16)f3.z; a[15] = (_Float16)f3.w;
        } else {
            #pragma unroll
            for (int j = 0; j < 8; ++j) {
                const int k = koff + j;
                a[j] = (k < 784) ? (_Float16)xr[k] : (_Float16)0.f;
            }
            #pragma unroll
            for (int j = 0; j < 8; ++j) {
                const int k = koff + 16 + j;
                a[8 + j] = (k < 784) ? (_Float16)xr[k] : (_Float16)0.f;
            }
        }

#if !FC1_TDM
        __syncthreads();
#endif
        // B fragments from f16 LDS (conflict-free), 4-deep register queue:
        // the loads for tile nt+4 issue right after the WMMA for tile nt,
        // landing 3 WMMAs before their consumer.
        const int klo2 = lo ? 0 : 8;
        auto loadB = [&](int nt) -> v16h {
            const _Float16* wrow = &wt[cur][(nt * 16 + lidx) * WT_STRH + klo2];
            const v8h b0 = *(const v8h*)wrow;
            const v8h b1 = *(const v8h*)(wrow + 16);
            v16h b;
            #pragma unroll
            for (int j = 0; j < 8; ++j) { b[j] = b0[j]; b[8 + j] = b1[j]; }
            return b;
        };
        v16h bq[4];
        #pragma unroll
        for (int d = 0; d < 4; ++d) bq[d] = loadB(d);
        #pragma unroll
        for (int nt = 0; nt < 16; ++nt) {
            acc[nt] = __builtin_amdgcn_wmma_f32_16x16x32_f16(
                false, a, false, bq[nt & 3], (short)0, acc[nt], false, false);
            if (nt + 4 < 16) bq[nt & 3] = loadB(nt + 4);
        }
    }

    // Epilogue: bias + relu, store f16 h1, accumulate column stats.
    #pragma unroll
    for (int nt = 0; nt < 16; ++nt) {
        const int   n    = nt * 16 + lidx;
        const float bias = b1[n];
        float ls = 0.f, lq = 0.f;
        #pragma unroll
        for (int r = 0; r < 8; ++r) {
            float v = acc[nt][r] + bias;
            v = v > 0.f ? v : 0.f;
            const int row = rowBase + r + (lo ? 0 : 8);
            h1[(size_t)row * 256 + n] = (_Float16)v;
            ls += v; lq += v * v;
        }
        atomicAdd(&csum[n], ls);
        atomicAdd(&csq[n], lq);
    }
    __syncthreads();
    if (tid < 256) {
        atomicAdd(&gsum[tid], csum[tid]);
        atomicAdd(&gsq[tid],  csq[tid]);
    }
}

// ---------------------------------------------------------------------------
// Kernel 2: finalize BN -> per-column scale/shift
// ---------------------------------------------------------------------------
__global__ void k_bn(const float* __restrict__ gsum, const float* __restrict__ gsq,
                     const float* __restrict__ gamma, const float* __restrict__ beta,
                     float* __restrict__ scale, float* __restrict__ shift) {
    const int t = threadIdx.x;
    const float inv  = 1.f / (float)BATCH;
    const float mean = gsum[t] * inv;
    const float var  = gsq[t] * inv - mean * mean;
    const float sc   = gamma[t] * rsqrtf(var + 1e-5f);
    scale[t] = sc;
    shift[t] = beta[t] - mean * sc;
}

// ---------------------------------------------------------------------------
// Kernel 3 helpers: fused Linear+ReLU+dropout layer, WMMA-tiled, LDS<->LDS.
// Flattened (kt,nt) loop; B fragments (global/L2) pipelined through a 4-deep
// register queue with in-place reload; A prefetched across kt from LDS.
// ---------------------------------------------------------------------------
template <int NIN, int NOUT>
__device__ __forceinline__ void layer_compute(const _Float16* __restrict__ src,
                                              const _Float16* __restrict__ Wp,
                                              v8f* acc, const int wv, const int lane) {
    constexpr int KT   = (NIN + 31) / 32;
    constexpr int KPAD = KT * 32;
    constexpr int NTC  = (NOUT + 15) / 16;
    constexpr int TOT  = KT * NTC;
    constexpr int DEP  = (TOT >= 4) ? 4 : ((TOT >= 2) ? 2 : 1);
    const int lo   = (lane < 16) ? 1 : 0;
    const int lidx = lane & 15;

    auto loadA = [&](int kt) -> v16h {
        const _Float16* srow = src + (wv * 16 + lidx) * ACT_STR + kt * 32 + (lo ? 0 : 16);
        const v8h a0 = *(const v8h*)srow;
        const v8h a1 = *(const v8h*)(srow + 8);
        v16h a;
        #pragma unroll
        for (int j = 0; j < 8; ++j) { a[j] = a0[j]; a[8 + j] = a1[j]; }
        return a;
    };
    auto loadB = [&](int t) -> v16h {
        const int kt = t / NTC;
        const int nt = t - kt * NTC;
        const int n  = nt * 16 + lidx;
        const int kb = kt * 32 + (lo ? 0 : 16);
        return *(const v16h*)(Wp + (size_t)n * KPAD + kb);   // 32B aligned
    };

    v16h bq[DEP];
    #pragma unroll
    for (int d = 0; d < DEP; ++d) bq[d] = loadB(d < TOT ? d : 0);

    v16h acur = loadA(0);
    v16h anext;

    #pragma unroll
    for (int t = 0; t < TOT; ++t) {
        const int kt = t / NTC;
        const int nt = t - kt * NTC;
        if (nt == 0 && kt + 1 < KT) anext = loadA(kt + 1);
        acc[nt] = __builtin_amdgcn_wmma_f32_16x16x32_f16(
            false, acur, false, bq[t % DEP], (short)0, acc[nt], false, false);
        if (t + DEP < TOT) bq[t % DEP] = loadB(t + DEP);
        if (nt == NTC - 1 && kt + 1 < KT) acur = anext;
    }
}

template <int NIN, int NOUT>
__device__ __forceinline__ void layer_fused(const _Float16* __restrict__ src,
                                            _Float16* __restrict__ dst,
                                            const _Float16* __restrict__ Wp,
                                            const float* __restrict__ Bp,
                                            const int layerIdx, const int gRow0,
                                            const int wv, const int lane) {
    constexpr int NTC = (NOUT + 15) / 16;
    constexpr int NTS = (NTC < 2) ? 2 : NTC;   // zero-pad to >=32 cols for next K chunk
    const int lo   = (lane < 16) ? 1 : 0;
    const int lidx = lane & 15;

    const v8f zerov = {0.f, 0.f, 0.f, 0.f, 0.f, 0.f, 0.f, 0.f};
    v8f acc[NTC];
    #pragma unroll
    for (int nt = 0; nt < NTC; ++nt) acc[nt] = zerov;

    layer_compute<NIN, NOUT>(src, Wp, acc, wv, lane);

    #pragma unroll
    for (int nt = 0; nt < NTS; ++nt) {
        const int n = nt * 16 + lidx;
        float bias = 0.f;
        if (nt < NTC && ((NOUT % 16 == 0) || n < NOUT)) bias = Bp[n];
        #pragma unroll
        for (int r = 0; r < 8; ++r) {
            const int row = wv * 16 + r + (lo ? 0 : 8);
            float v = 0.f;
            if (nt < NTC) {
                v = acc[nt][r] + bias;
                v = v > 0.f ? v : 0.f;
                if (!((NOUT % 16 == 0) || n < NOUT)) v = 0.f;
                // dropout(0.5): deterministic hash keep/drop, x2 rescale
                const uint32_t h = mixh((uint32_t)(gRow0 + row) * 0x9E3779B9u
                                      + (uint32_t)n * 0x85EBCA6Bu
                                      + (uint32_t)layerIdx * 0xC2B2AE35u);
                v = (h & 0x80000000u) ? v * 2.f : 0.f;
            }
            dst[row * ACT_STR + n] = (_Float16)v;
        }
    }
}

__device__ __forceinline__ void layer_final(const _Float16* __restrict__ src,
                                            const _Float16* __restrict__ Wp,
                                            const float* __restrict__ Bp,
                                            float* __restrict__ out,
                                            const int gRow0, const int wv,
                                            const int lane) {
    const int lo   = (lane < 16) ? 1 : 0;
    const int lidx = lane & 15;

    const v8f zerov = {0.f, 0.f, 0.f, 0.f, 0.f, 0.f, 0.f, 0.f};
    v8f acc[16];
    #pragma unroll
    for (int nt = 0; nt < 16; ++nt) acc[nt] = zerov;

    layer_compute<256, 256>(src, Wp, acc, wv, lane);

    #pragma unroll
    for (int nt = 0; nt < 16; ++nt) {
        const float bias = Bp[nt * 16 + lidx];
        #pragma unroll
        for (int r = 0; r < 8; ++r) acc[nt][r] += bias;
    }

    // log_softmax over 256 cols of each row: in-lane over 16 N-tiles, then
    // 16-lane shfl_xor reductions (lane halves hold independent rows).
    #pragma unroll
    for (int r = 0; r < 8; ++r) {
        float m = -3.0e38f;
        #pragma unroll
        for (int nt = 0; nt < 16; ++nt) m = fmaxf(m, acc[nt][r]);
        m = fmaxf(m, __shfl_xor(m, 1, 32));
        m = fmaxf(m, __shfl_xor(m, 2, 32));
        m = fmaxf(m, __shfl_xor(m, 4, 32));
        m = fmaxf(m, __shfl_xor(m, 8, 32));
        float s = 0.f;
        #pragma unroll
        for (int nt = 0; nt < 16; ++nt) s += __expf(acc[nt][r] - m);
        s += __shfl_xor(s, 1, 32);
        s += __shfl_xor(s, 2, 32);
        s += __shfl_xor(s, 4, 32);
        s += __shfl_xor(s, 8, 32);
        const float lse = m + __logf(s);
        const int   row = gRow0 + wv * 16 + r + (lo ? 0 : 8);
        float* orow = out + (size_t)row * 256;
        #pragma unroll
        for (int nt = 0; nt < 16; ++nt) orow[nt * 16 + lidx] = acc[nt][r] - lse;
    }
}

// ---------------------------------------------------------------------------
// Kernel 3: BN affine -> fc2..fc19 (fused, LDS-resident) -> fc20 + log_softmax
// ---------------------------------------------------------------------------
__global__ __launch_bounds__(256) void k_tail(const _Float16* __restrict__ h1,
                                              const float* __restrict__ scale,
                                              const float* __restrict__ shift,
                                              Params p, float* __restrict__ out) {
    __shared__ _Float16 act[2][MBLK * ACT_STR];   // 2 x 66 KB ping-pong
    __shared__ float scl[256], sft[256];

    const int tid     = threadIdx.x;
    const int lane    = tid & 31;
    const int wv      = tid >> 5;
    const int rowBase = blockIdx.x * MBLK;

    if (tid < 256) { scl[tid] = scale[tid]; sft[tid] = shift[tid]; }
    __syncthreads();

    // Stage 0: load h1 block, apply BatchNorm affine, f16 into LDS.
    for (int idx = tid; idx < MBLK * 256; idx += 256) {
        const int r = idx >> 8, n = idx & 255;
        const float v = (float)h1[(size_t)(rowBase + r) * 256 + n];
        act[0][r * ACT_STR + n] = (_Float16)(v * scl[n] + sft[n]);
    }
    __syncthreads();

    layer_fused<256, 256>(act[0], act[1], p.W16[1],  p.Bv[1],  1,  rowBase, wv, lane); __syncthreads();
    layer_fused<256, 128>(act[1], act[0], p.W16[2],  p.Bv[2],  2,  rowBase, wv, lane); __syncthreads();
    layer_fused<128,  64>(act[0], act[1], p.W16[3],  p.Bv[3],  3,  rowBase, wv, lane); __syncthreads();
    layer_fused< 64,  32>(act[1], act[0], p.W16[4],  p.Bv[4],  4,  rowBase, wv, lane); __syncthreads();
    layer_fused< 32,  16>(act[0], act[1], p.W16[5],  p.Bv[5],  5,  rowBase, wv, lane); __syncthreads();
    layer_fused< 16,   8>(act[1], act[0], p.W16[6],  p.Bv[6],  6,  rowBase, wv, lane); __syncthreads();
    layer_fused<  8,   4>(act[0], act[1], p.W16[7],  p.Bv[7],  7,  rowBase, wv, lane); __syncthreads();
    layer_fused<  4,   2>(act[1], act[0], p.W16[8],  p.Bv[8],  8,  rowBase, wv, lane); __syncthreads();
    layer_fused<  2,   2>(act[0], act[1], p.W16[9],  p.Bv[9],  9,  rowBase, wv, lane); __syncthreads();
    layer_fused<  2,   2>(act[1], act[0], p.W16[10], p.Bv[10], 10, rowBase, wv, lane); __syncthreads();
    layer_fused<  2,   4>(act[0], act[1], p.W16[11], p.Bv[11], 11, rowBase, wv, lane); __syncthreads();
    layer_fused<  4,   8>(act[1], act[0], p.W16[12], p.Bv[12], 12, rowBase, wv, lane); __syncthreads();
    layer_fused<  8,  16>(act[0], act[1], p.W16[13], p.Bv[13], 13, rowBase, wv, lane); __syncthreads();
    layer_fused< 16,  32>(act[1], act[0], p.W16[14], p.Bv[14], 14, rowBase, wv, lane); __syncthreads();
    layer_fused< 32,  64>(act[0], act[1], p.W16[15], p.Bv[15], 15, rowBase, wv, lane); __syncthreads();
    layer_fused< 64, 128>(act[1], act[0], p.W16[16], p.Bv[16], 16, rowBase, wv, lane); __syncthreads();
    layer_fused<128, 256>(act[0], act[1], p.W16[17], p.Bv[17], 17, rowBase, wv, lane); __syncthreads();
    layer_fused<256, 256>(act[1], act[0], p.W16[18], p.Bv[18], 18, rowBase, wv, lane); __syncthreads();

    layer_final(act[0], p.W16[19], p.Bv[19], out, rowBase, wv, lane);
}

// ---------------------------------------------------------------------------
extern "C" void kernel_launch(void* const* d_in, const int* in_sizes, int n_in,
                              void* d_out, int out_size, void* d_ws, size_t ws_size,
                              hipStream_t stream) {
    (void)in_sizes; (void)n_in; (void)out_size; (void)ws_size;

    static const int SIZES[21] =
        {784, 256, 256, 128, 64, 32, 16, 8, 4, 2, 2, 2, 4, 8, 16, 32, 64, 128, 256, 256, 256};

    const float* x = (const float*)d_in[0];
    const float* gamma = (const float*)d_in[41];
    const float* beta  = (const float*)d_in[42];
    float* out = (float*)d_out;

    // Workspace layout:
    //   [h1 f16: 64K x 256 = 32 MB][stats: 4 x 256 f32][padded f16 weights ~1 MB]
    _Float16* h1    = (_Float16*)d_ws;
    float*    stats = (float*)((char*)d_ws + (size_t)BATCH * 256 * sizeof(_Float16));
    float* gsum  = stats;
    float* gsq   = stats + 256;
    float* scale = stats + 512;
    float* shift = stats + 768;
    _Float16* wbase = (_Float16*)(stats + 1024);

    PrepPar q;
    Params  p;
    size_t off = 0;
    for (int i = 0; i < 20; ++i) {
        const int kp = ((SIZES[i]     + 31) >> 5) << 5;
        const int np = ((SIZES[i + 1] + 15) >> 4) << 4;
        q.W[i]   = (const float*)d_in[1 + i];
        q.D[i]   = wbase + off;
        p.W16[i] = wbase + off;
        p.Bv[i]  = (const float*)d_in[21 + i];
        off += (size_t)np * kp;
    }

    hipLaunchKernelGGL(k_prep, dim3(NLAYERS + 1), dim3(256), 0, stream, q, gsum, gsq);
    hipLaunchKernelGGL(k_fc1,  dim3(BATCH / MBLK), dim3(256), 0, stream,
                       x, p.W16[0], p.Bv[0], h1, gsum, gsq);
    hipLaunchKernelGGL(k_bn,   dim3(1),   dim3(256), 0, stream,
                       gsum, gsq, gamma, beta, scale, shift);
    hipLaunchKernelGGL(k_tail, dim3(BATCH / MBLK), dim3(256), 0, stream,
                       h1, scale, shift, p, out);
}
